// Protonet_37735582663367
// MI455X (gfx1250) — matrix operand order repaired and checked by reading, not compiled, not run on hardware
//
#include <hip/hip_runtime.h>

// ---------------------------------------------------------------------------
// ProtoNet loss for MI455X (gfx1250). Three WMMA GEMMs (bf16 in, f32 acc):
//   K1: z = [xs;xq] @ W            (17664 x 1024 x 128)
//   K4: dot = zq @ proto^T         (16384 x 128 x 256)  -> d2
//   K5: sim = a @ a^T per class    (256 batches of 64 x 256 x 64)
// bf16 conversion done once on LDS store (shared by all 8 waves); LDS rows
// padded (+1 dword) to kill bank conflicts on the strided A-fragment reads.
// ---------------------------------------------------------------------------

typedef __attribute__((ext_vector_type(16))) __bf16 v16bf;
typedef __attribute__((ext_vector_type(8)))  float  v8f;

#define C_CLS 256
#define S_SUP 5
#define Q_QRY 64
#define DIN   1024
#define ZD    128
#define NS    (C_CLS * S_SUP)   // 1280 support rows
#define NQ    (C_CLS * Q_QRY)   // 16384 query rows
#define NTOT  (NS + NQ)         // 17664
#define KC    128               // K-chunk staged per barrier in k_gemm_z
#define LDW   (KC / 2 + 1)      // padded LDS row stride in dwords (65)

union BF16x16 { v16bf v; unsigned int u[8]; };

__device__ __forceinline__ unsigned short f2bf(float f) {
  unsigned int u = __float_as_uint(f);
  u += 0x7FFFu + ((u >> 16) & 1u);      // round-to-nearest-even
  return (unsigned short)(u >> 16);
}
__device__ __forceinline__ unsigned int pack2bf(float lo, float hi) {
#if __has_builtin(__builtin_amdgcn_cvt_pk_bf16_f32)
  typedef __attribute__((ext_vector_type(2))) __bf16 v2bf;
  union { v2bf v; unsigned int u; } cv;
  cv.v = __builtin_amdgcn_cvt_pk_bf16_f32(lo, hi);
  return cv.u;
#else
  return (unsigned int)f2bf(lo) | ((unsigned int)f2bf(hi) << 16);
#endif
}
__device__ __forceinline__ float wave_sum(float v) {
  #pragma unroll
  for (int o = 16; o > 0; o >>= 1) v += __shfl_xor(v, o, 32);
  return v;
}

// --- K0: convert W to bf16, zero scalar accumulators -----------------------
__global__ __launch_bounds__(256) void k_init(const float* __restrict__ W,
                                              unsigned short* __restrict__ Wbf,
                                              float* __restrict__ accum) {
  int i = blockIdx.x * 256 + threadIdx.x;
  if (i < DIN * ZD) Wbf[i] = f2bf(W[i]);
  if (i < 2) accum[i] = 0.f;
}

// --- K1: z = x @ W via wmma bf16. Block: 16 rows x 128 cols, 8 waves -------
// Double-buffered bf16 LDS tiles (16 x KC), converted once on store.
__global__ __launch_bounds__(256) void k_gemm_z(const float* __restrict__ xs,
                                                const float* __restrict__ xq,
                                                const unsigned int* __restrict__ Wb, // bf16 pairs
                                                float* __restrict__ z,
                                                unsigned short* __restrict__ zbf) {
  __shared__ unsigned int xt[2][16 * LDW];  // padded: bank = (row+col)%64
  const int tid = threadIdx.x, lane = tid & 31, wave = tid >> 5;
  const int row0 = blockIdx.x * 16;
  const int nw = wave * 16;                 // this wave's N offset
  const int am = lane & 15, kg = lane >> 4; // A-frag row, lane half
  // a 16-row tile never straddles the xs/xq boundary (NS = 80*16)
  const float* xbase = (row0 < NS) ? (xs + (size_t)row0 * DIN)
                                   : (xq + (size_t)(row0 - NS) * DIN);
  v8f acc = {};

  auto stage = [&](int k0, int buf) {
    #pragma unroll
    for (int p = tid; p < 16 * (KC / 2); p += 256) {
      int r = p >> 6, cpair = p & 63;       // KC/2 = 64 pairs per row
      float2 f = ((const float2*)(xbase + (size_t)r * DIN + k0))[cpair];
      xt[buf][r * LDW + cpair] = pack2bf(f.x, f.y);
      if (k0 + KC < DIN)
        __builtin_prefetch(xbase + (size_t)r * DIN + k0 + KC + 2 * cpair, 0, 1);
    }
  };

  stage(0, 0);
  for (int k0 = 0; k0 < DIN; k0 += KC) {
    __syncthreads();
    const int buf = (k0 / KC) & 1;
    if (k0 + KC < DIN) stage(k0 + KC, buf ^ 1);   // overlap with WMMA below
    const unsigned int* xrow = &xt[buf][am * LDW];
    #pragma unroll
    for (int kk = 0; kk < KC; kk += 32) {
      BF16x16 a, b;
      #pragma unroll
      for (int v = 0; v < 8; ++v) {         // ISA 16-bit A 16x32 layout
        int kkv = kk + ((v >> 2) << 4) + (kg << 3) + ((v & 3) << 1);
        a.u[v] = xrow[kkv >> 1];
      }
      const unsigned short* Ws = (const unsigned short*)Wb;  // B: N on lanes, K pairs on VGPRs (same K order as A)
      #pragma unroll
      for (int v = 0; v < 8; ++v) {
        int krow = k0 + kk + ((v >> 2) << 4) + (kg << 3) + ((v & 3) << 1);
        b.u[v] = (unsigned int)Ws[krow * ZD + nw + am] |
                 ((unsigned int)Ws[(krow + 1) * ZD + nw + am] << 16);
      }
      acc = __builtin_amdgcn_wmma_f32_16x16x32_bf16(false, a.v, false, b.v,
                                                    (short)0, acc, false, false);
    }
  }
  const int n = lane & 15;
  #pragma unroll
  for (int v = 0; v < 8; ++v) {             // C/D layout: M = v + 8*laneHalf
    int grow = row0 + v + kg * 8;
    float val = acc[v];
    z[(size_t)grow * ZD + nw + n]   = val;
    zbf[(size_t)grow * ZD + nw + n] = f2bf(val);
  }
}

// --- K2: proto mean, proto^T (bf16) and ||proto||^2 ------------------------
__global__ __launch_bounds__(128) void k_proto(const float* __restrict__ z,
                                               unsigned short* __restrict__ protoTbf,
                                               float* __restrict__ proto_sq) {
  __shared__ float red[128];
  const int c = blockIdx.x, n = threadIdx.x;
  float p = 0.f;
  #pragma unroll
  for (int s = 0; s < S_SUP; ++s) p += z[(size_t)(c * S_SUP + s) * ZD + n];
  p *= (1.f / S_SUP);
  protoTbf[n * C_CLS + c] = f2bf(p);        // stored transposed: [k][n]
  red[n] = p * p;
  __syncthreads();
  #pragma unroll
  for (int off = 64; off > 0; off >>= 1) {
    if (n < off) red[n] += red[n + off];
    __syncthreads();
  }
  if (n == 0) proto_sq[c] = red[0];
}

// --- K3: ||zq||^2 per query row (one wave per row) -------------------------
__global__ __launch_bounds__(256) void k_zqsq(const float* __restrict__ zq,
                                              float* __restrict__ zq_sq) {
  const int wid  = (blockIdx.x * 256 + threadIdx.x) >> 5;
  const int lane = threadIdx.x & 31;
  if (wid >= NQ) return;
  float s = 0.f;
  #pragma unroll
  for (int j = lane; j < ZD; j += 32) {
    float v = zq[(size_t)wid * ZD + j];
    s += v * v;
  }
  s = wave_sum(s);
  if (lane == 0) zq_sq[wid] = s;
}

// --- K4: d2 = ||zq||^2 + ||proto||^2 - 2 zq@proto^T  (WMMA) ----------------
__global__ __launch_bounds__(256) void k_d2(const unsigned int* __restrict__ zqb, // 16384x128 bf16 pairs
                                            const unsigned int* __restrict__ ptb, // 128x256 bf16 pairs
                                            const float* __restrict__ zq_sq,
                                            const float* __restrict__ proto_sq,
                                            float* __restrict__ d2) {
  const int lane = threadIdx.x & 31, wave = threadIdx.x >> 5;
  const int row0 = blockIdx.x * 16;
  const int n0   = blockIdx.y * 128 + wave * 16;
  const int m = row0 + (lane & 15), kg = lane >> 4;
  v8f acc = {};
  #pragma unroll
  for (int k0 = 0; k0 < ZD; k0 += 32) {
    BF16x16 a, b;
    #pragma unroll
    for (int v = 0; v < 8; ++v) {
      int kk = k0 + ((v >> 2) << 4) + (kg << 3) + ((v & 3) << 1);
      a.u[v] = zqb[(size_t)m * (ZD / 2) + (kk >> 1)];
    }
    const unsigned short* ptbs = (const unsigned short*)ptb;
    #pragma unroll
    for (int v = 0; v < 8; ++v) {
      int kr = k0 + ((v >> 2) << 4) + (kg << 3) + ((v & 3) << 1);
      b.u[v] = (unsigned int)ptbs[kr * C_CLS + n0 + (lane & 15)] |
               ((unsigned int)ptbs[(kr + 1) * C_CLS + n0 + (lane & 15)] << 16);
    }
    acc = __builtin_amdgcn_wmma_f32_16x16x32_bf16(false, a.v, false, b.v,
                                                  (short)0, acc, false, false);
  }
  const int nn = lane & 15;
  #pragma unroll
  for (int v = 0; v < 8; ++v) {
    int r = row0 + v + kg * 8, cc = n0 + nn;
    d2[(size_t)r * C_CLS + cc] = zq_sq[r] + proto_sq[cc] - 2.f * acc[v];
  }
}

// --- K5: per-class cosine-sim GEMM. delete-col == zero-col trick. ----------
// Block c: a = normalize(d2[c-block] with col c zeroed), sim = a a^T (64x64,
// K=256 via 8 bf16 wmma steps), sum strict upper triangle -> atomicAdd.
#define AW 258                               // padded row stride (shorts)
__global__ __launch_bounds__(256) void k_cos(const float* __restrict__ d2,
                                             float* __restrict__ accum) {
  __shared__ unsigned short abf[64 * AW];    // ~32.25 KB normalized bf16 tile
  __shared__ float invn[64];
  const int c = blockIdx.x;
  const int tid = threadIdx.x, lane = tid & 31, wave = tid >> 5;

  // phase 1: row inverse norms (col c excluded)
  for (int qr = wave; qr < 64; qr += 8) {
    const size_t gq = (size_t)(c * Q_QRY + qr) * C_CLS;
    float s = 0.f;
    #pragma unroll
    for (int j = lane; j < C_CLS; j += 32) {
      float v = (j == c) ? 0.f : d2[gq + j];
      s += v * v;
    }
    s = wave_sum(s);
    if (lane == 0) invn[qr] = 1.f / fmaxf(sqrtf(s), 1e-8f);
  }
  __syncthreads();

  // phase 2: normalized bf16 tile into LDS (padded rows)
  for (int e = tid; e < 64 * 256; e += 256) {
    int qr = e >> 8, j = e & 255;
    float v = (j == c) ? 0.f : d2[(size_t)(c * Q_QRY + qr) * C_CLS + j];
    abf[qr * AW + j] = f2bf(v * invn[qr]);
  }
  __syncthreads();

  // phase 3: 4x4 grid of 16x16 output tiles; 2 tiles per wave
  const unsigned int* a32 = (const unsigned int*)abf;
  float part = 0.f;
  #pragma unroll
  for (int t = wave * 2; t < wave * 2 + 2; ++t) {
    const int mt = (t >> 2) * 16, nt = (t & 3) * 16;
    const int m = mt + (lane & 15), kg = lane >> 4;
    v8f acc = {};
    #pragma unroll
    for (int k0 = 0; k0 < 256; k0 += 32) {
      BF16x16 a, b;
      #pragma unroll
      for (int v = 0; v < 8; ++v) {
        int kk = k0 + ((v >> 2) << 4) + (kg << 3) + ((v & 3) << 1);
        a.u[v] = a32[m * (AW / 2) + (kk >> 1)];
      }
      const int n = nt + (lane & 15);
      #pragma unroll
      for (int v = 0; v < 8; ++v) {  // B[k][n] = a[n][k]
        int kr = k0 + ((v >> 2) << 4) + (kg << 3) + ((v & 3) << 1);
        b.u[v] = a32[n * (AW / 2) + (kr >> 1)];
      }
      acc = __builtin_amdgcn_wmma_f32_16x16x32_bf16(false, a.v, false, b.v,
                                                    (short)0, acc, false, false);
    }
    const int nn = nt + (lane & 15);
    #pragma unroll
    for (int v = 0; v < 8; ++v) {
      int mm = mt + v + kg * 8;
      if (nn > mm) part += acc[v];      // strict upper triangle (q < q')
    }
  }
  part = wave_sum(part);
  if (lane == 0) atomicAdd(&accum[0], part);
}

// --- K6: accuracy = mean(argmin_c d2[q,c] == class(q)) ---------------------
__global__ __launch_bounds__(256) void k_acc(const float* __restrict__ d2,
                                             float* __restrict__ accum) {
  const int wid  = (blockIdx.x * 256 + threadIdx.x) >> 5;
  const int lane = threadIdx.x & 31;
  if (wid >= NQ) return;
  float best = 3.4e38f; int bidx = C_CLS;
  #pragma unroll
  for (int j = lane; j < C_CLS; j += 32) {
    float v = d2[(size_t)wid * C_CLS + j];
    if (v < best || (v == best && j < bidx)) { best = v; bidx = j; }
  }
  #pragma unroll
  for (int off = 16; off > 0; off >>= 1) {
    float ov = __shfl_xor(best, off, 32);
    int   oi = __shfl_xor(bidx, off, 32);
    if (ov < best || (ov == best && oi < bidx)) { best = ov; bidx = oi; }
  }
  if (lane == 0 && bidx == (wid >> 6)) atomicAdd(&accum[1], 1.f);
}

// --- K7: finalize ----------------------------------------------------------
__global__ void k_final(const float* __restrict__ accum, float* __restrict__ out) {
  if (blockIdx.x == 0 && threadIdx.x == 0) {
    out[0] = accum[0] / 516096.f;   // C * Q*(Q-1)/2
    out[1] = accum[1] / 16384.f;    // C * Q
  }
}

extern "C" void kernel_launch(void* const* d_in, const int* in_sizes, int n_in,
                              void* d_out, int out_size, void* d_ws, size_t ws_size,
                              hipStream_t stream) {
  const float* xs = (const float*)d_in[0];   // (256,5,1024)
  const float* xq = (const float*)d_in[1];   // (256,64,1024)
  const float* W  = (const float*)d_in[2];   // (1024,128)

  char* ws = (char*)d_ws;
  size_t off = 0;
  float*          z     = (float*)(ws + off);          off += (size_t)NTOT * ZD * 4;
  unsigned short* zbf   = (unsigned short*)(ws + off); off += (size_t)NTOT * ZD * 2;
  unsigned short* Wbf   = (unsigned short*)(ws + off); off += (size_t)DIN * ZD * 2;
  unsigned short* ptb   = (unsigned short*)(ws + off); off += (size_t)ZD * C_CLS * 2;
  float*          psq   = (float*)(ws + off);          off += (size_t)C_CLS * 4;
  float*          qsq   = (float*)(ws + off);          off += (size_t)NQ * 4;
  float*          d2    = (float*)(ws + off);          off += (size_t)NQ * C_CLS * 4;
  float*          accum = (float*)(ws + off);          off += 256;

  k_init<<<(DIN * ZD + 255) / 256, 256, 0, stream>>>(W, Wbf, accum);
  k_gemm_z<<<NTOT / 16, 256, 0, stream>>>(xs, xq, (const unsigned int*)Wbf, z, zbf);
  k_proto<<<C_CLS, 128, 0, stream>>>(z, ptb, psq);
  k_zqsq<<<NQ / 8, 256, 0, stream>>>(z + (size_t)NS * ZD, qsq);
  k_d2<<<dim3(NQ / 16, 2), 256, 0, stream>>>(
      (const unsigned int*)(zbf + (size_t)NS * ZD),
      (const unsigned int*)ptb, qsq, psq, d2);
  k_cos<<<C_CLS, 256, 0, stream>>>(d2, accum);
  k_acc<<<NQ / 8, 256, 0, stream>>>(d2, accum);
  k_final<<<1, 32, 0, stream>>>(accum, (float*)d_out);
}
